// SobaMonarch_4784593567745
// MI455X (gfx1250) — matrix operand order repaired and checked by reading, not compiled, not run on hardware
//
#include <hip/hip_runtime.h>

// Monarch factor attention for MI455X (gfx1250, wave32).
// bf16 WMMA (v_wmma_f32_16x16x32_bf16) for all GEMMs, fp32 VALU for
// normalization / projection / gradient-step math.
// All WMMA fragment loads are K-contiguous 16B LDS vector loads:
// A operands stored row-major (MxK), B operands stored N-major ([n][k]).

#define SEQ    2048
#define LB     32     // L = S / block_size
#define BSZ    64     // block size
#define DDIM   64     // head dim
#define NH     12
#define NBH    48     // B * H
#define NSTEPS 4

typedef __attribute__((ext_vector_type(16))) __bf16 v16bf;
typedef __attribute__((ext_vector_type(8)))  __bf16 v8bf;
typedef __attribute__((ext_vector_type(4)))  __bf16 v4bf;
typedef __attribute__((ext_vector_type(8)))  float  v8f;

__device__ __forceinline__ float softplusf(float x) {
  return fmaxf(x, 0.f) + log1pf(__expf(-fabsf(x)));
}

__device__ __forceinline__ v16bf cat16(v8bf lo, v8bf hi) {
  return __builtin_shufflevector(lo, hi, 0, 1, 2, 3, 4, 5, 6, 7, 8, 9, 10, 11,
                                 12, 13, 14, 15);
}

// ---------------- WMMA 16x16x32 bf16 fragment helpers (wave32) -------------
// A fragment: logical A(m,k), source row-major MxK with leading dim ld.
// Lane (m,h): a[0..7] = A(m, 8h+0..7), a[8..15] = A(m, 16+8h+0..7).
__device__ __forceinline__ v16bf load_a(const __bf16* src, int ld, int lane) {
  int m = lane & 15, h = lane >> 4;
  const __bf16* p = src + m * ld + 8 * h;
  v8bf lo = *(const v8bf*)(p);
  v8bf hi = *(const v8bf*)(p + 16);
  return cat16(lo, hi);
}
// B fragment: logical B(k,n), source N-major: element (k,n) at src[n*ld + k].
// Lane (n,h): b[0..7] = B(8h+0..7, n), b[8..15] = B(16+8h+0..7, n).
__device__ __forceinline__ v16bf load_b(const __bf16* src, int ld, int lane) {
  int n = lane & 15, h = lane >> 4;
  const __bf16* p = src + n * ld + 8 * h;
  v8bf lo = *(const v8bf*)(p);
  v8bf hi = *(const v8bf*)(p + 16);
  return cat16(lo, hi);
}
// C/D 16x16 f32: VGPR r -> M = r + 8*half, N = lane&15.
__device__ __forceinline__ void store_c(float* dst, int ld, v8f c, int lane) {
  int n = lane & 15, h = lane >> 4;
#pragma unroll
  for (int r = 0; r < 8; ++r) dst[(r + 8 * h) * ld + n] = c[r];
}
__device__ __forceinline__ v8f wmma_bf16(v16bf a, v16bf b, v8f c) {
  return __builtin_amdgcn_wmma_f32_16x16x32_bf16(false, a, false, b, (short)0,
                                                 c, false, false);
}

// ---------------------------------------------------------------------------
__global__ void k_init(float* __restrict__ lp, float* __restrict__ rp) {
  size_t i = (size_t)blockIdx.x * blockDim.x + threadIdx.x;
  const size_t nlp = (size_t)NBH * BSZ * LB * LB;
  const size_t nrp = (size_t)NBH * LB * BSZ * BSZ;
  if (i < nlp) lp[i] = 0.17677669529663687f;  // 1/sqrt(32)
  if (i < nrp) rp[i] = 0.125f;                // 1/sqrt(64)
}

// Per (bh,k): normalize rp row-wise -> right, r2; KR = right(64x64) @ M(64x64).
// Also used for the final Y = right_final @ V (r2out == nullptr).
__global__ void k_right_gemm(const float* __restrict__ rp,
                             const float* __restrict__ mat,  // key or value
                             float* __restrict__ outKR,      // (bh,k,j,v)
                             float* __restrict__ r2out) {    // (bh,k,j) or null
  int blk = blockIdx.x;  // bh*LB + k
  int tid = threadIdx.x, lane = tid & 31, wave = tid >> 5;
  __shared__ __align__(16) float  sRP[BSZ][BSZ];
  __shared__ __align__(16) __bf16 sRight[BSZ][BSZ];   // A: row-major (j,i)
  __shared__ __align__(16) __bf16 sMt[DDIM][BSZ];     // B N-major: [v][i]

  const float* rpk = rp + (size_t)blk * BSZ * BSZ;
  const float* Mk  = mat + (size_t)blk * BSZ * DDIM;
  for (int idx = tid; idx < BSZ * BSZ / 4; idx += 256) {
    int e = 4 * idx;
    *(float4*)&sRP[e >> 6][e & 63] = *(const float4*)&rpk[e];
  }
  for (int idx = tid; idx < BSZ * DDIM / 4; idx += 256) {
    int e = 4 * idx;
    int i = e >> 6, v = e & 63;
    float4 m4 = *(const float4*)&Mk[e];
    sMt[v + 0][i] = (__bf16)m4.x;
    sMt[v + 1][i] = (__bf16)m4.y;
    sMt[v + 2][i] = (__bf16)m4.z;
    sMt[v + 3][i] = (__bf16)m4.w;
  }
  __syncthreads();

  if (tid < BSZ) {
    int j = tid;
    const float4* row = (const float4*)&sRP[j][0];
    float ss = 0.f;
    for (int i = 0; i < BSZ / 4; ++i) {
      float4 v = row[i];
      ss += v.x * v.x + v.y * v.y + v.z * v.z + v.w * v.w;
    }
    float n = sqrtf(ss);
    float inv = n > 0.f ? 1.f / n : 1.f;  // safe_normalize
    float sr2 = 0.f;
    for (int i = 0; i < BSZ / 4; ++i) {
      float4 v = row[i];
      float r0 = v.x * inv, r1 = v.y * inv, r2v = v.z * inv, r3 = v.w * inv;
      float t0 = r0 * r0, t1 = r1 * r1, t2 = r2v * r2v, t3 = r3 * r3;
      v4bf pk = {(__bf16)t0, (__bf16)t1, (__bf16)t2, (__bf16)t3};
      *(v4bf*)&sRight[j][4 * i] = pk;
      sr2 += t0 * t0 + t1 * t1 + t2 * t2 + t3 * t3;
    }
    if (r2out) r2out[(size_t)blk * BSZ + j] = sr2;
  }
  __syncthreads();

  float* dstk = outKR + (size_t)blk * BSZ * DDIM;
#pragma unroll
  for (int t = wave; t < 16; t += 8) {  // 4x4 tiles, 8 waves
    int mt = t >> 2, nt = t & 3;
    v8f acc = {};
#pragma unroll
    for (int ks = 0; ks < 2; ++ks) {
      v16bf a = load_a(&sRight[mt * 16][ks * 32], BSZ, lane);
      v16bf b = load_b(&sMt[nt * 16][ks * 32], BSZ, lane);
      acc = wmma_bf16(a, b, acc);
    }
    store_c(&dstk[(mt * 16) * DDIM + nt * 16], DDIM, acc, lane);
  }
}

// Per (bh,j): normalize lp -> ls/left/l2; t2l = Q @ KR^T; LQ = left^T @ Q;
// dl with projection; lp -= eta_l * dl.
__global__ void k_left_step(float* __restrict__ lp,
                            const float* __restrict__ qg,
                            const float* __restrict__ KR,
                            const float* __restrict__ r2w,
                            const float* __restrict__ attn_scale,
                            const float* __restrict__ step_size, int step,
                            float* __restrict__ LQ,    // (bh,j,k,v)
                            float* __restrict__ l2w) { // (bh,j,k)
  int blk = blockIdx.x;  // bh*BSZ + j
  int bh = blk >> 6, j = blk & 63;
  int h = bh % NH;
  int tid = threadIdx.x, lane = tid & 31, wave = tid >> 5;
  __shared__ __align__(16) float  sLP[LB][LB];
  __shared__ __align__(16) float  sLS[LB][LB];
  __shared__ __align__(16) __bf16 sLeftT[LB][LB];   // A for LQ: [k][l]
  __shared__ __align__(16) __bf16 sQ[LB][DDIM];     // A for t2l: (l,v)
  __shared__ __align__(16) __bf16 sQt[DDIM][LB];    // B for LQ: [v][l]
  __shared__ __align__(16) __bf16 sKR[LB][DDIM];    // B for t2l (=KR^T N-major)
  __shared__ __align__(16) float  sT2[LB][LB];
  __shared__ float  sR2[LB];
  __shared__ float  sInv[LB];

  float qscale = softplusf(attn_scale[h]) * 0.125f;  // softplus(a)/sqrt(D)
  float* lpj = lp + (size_t)blk * LB * LB;
  for (int idx = tid; idx < LB * LB / 4; idx += 256) {
    int e = 4 * idx;
    *(float4*)&sLP[e >> 5][e & 31] = *(const float4*)&lpj[e];
  }
  for (int idx = tid; idx < LB * DDIM / 4; idx += 256) {
    int e = 4 * idx;
    int l = e >> 6, v = e & 63;
    float4 q4 = *(const float4*)&qg[((size_t)bh * SEQ + l * BSZ + j) * DDIM + v];
    __bf16 b0 = (__bf16)(q4.x * qscale), b1 = (__bf16)(q4.y * qscale);
    __bf16 b2 = (__bf16)(q4.z * qscale), b3 = (__bf16)(q4.w * qscale);
    v4bf pk = {b0, b1, b2, b3};
    *(v4bf*)&sQ[l][v] = pk;
    sQt[v + 0][l] = b0;
    sQt[v + 1][l] = b1;
    sQt[v + 2][l] = b2;
    sQt[v + 3][l] = b3;
  }
  for (int idx = tid; idx < LB * DDIM / 4; idx += 256) {
    int e = 4 * idx;
    int k = e >> 6, v = e & 63;
    float4 kr4 =
        *(const float4*)&KR[(((size_t)bh * LB + k) * BSZ + j) * DDIM + v];
    v4bf pk = {(__bf16)kr4.x, (__bf16)kr4.y, (__bf16)kr4.z, (__bf16)kr4.w};
    *(v4bf*)&sKR[k][v] = pk;
  }
  if (tid < LB) sR2[tid] = r2w[((size_t)bh * LB + tid) * BSZ + j];
  __syncthreads();

  if (tid < LB) {
    int l = tid;
    const float4* row = (const float4*)&sLP[l][0];
    float ss = 0.f;
    for (int k = 0; k < LB / 4; ++k) {
      float4 v = row[k];
      ss += v.x * v.x + v.y * v.y + v.z * v.z + v.w * v.w;
    }
    float n = sqrtf(ss);
    float inv = n > 0.f ? 1.f / n : 1.f;
    sInv[l] = n > 0.f ? 1.f / n : 0.f;  // safe_inv_norm
    for (int k = 0; k < LB; ++k) {
      float lsv = sLP[l][k] * inv;
      sLS[l][k] = lsv;
      sLeftT[k][l] = (__bf16)(lsv * lsv);
    }
  }
  __syncthreads();

  if (tid < LB) {  // l2[k] = sum_l left^2
    int k = tid;
    float ss = 0.f;
    for (int l = 0; l < LB; ++l) {
      float lf = sLS[l][k] * sLS[l][k];
      ss += lf * lf;
    }
    l2w[(size_t)blk * LB + k] = ss;
  }
  if (wave < 4) {  // t2l (32x32) = Q(32x64) @ KR^T(64x32), 2x2 tiles
    int mt = wave >> 1, nt = wave & 1;
    v8f acc = {};
#pragma unroll
    for (int ks = 0; ks < 2; ++ks) {
      v16bf a = load_a(&sQ[mt * 16][ks * 32], DDIM, lane);
      v16bf b = load_b(&sKR[nt * 16][ks * 32], DDIM, lane);
      acc = wmma_bf16(a, b, acc);
    }
    store_c(&sT2[mt * 16][nt * 16], LB, acc, lane);
  }
  {  // LQ (32x64) = left^T(32x32) @ Q(32x64), 2x4 tiles, all 8 waves
    int mt = wave >> 2, nt = wave & 3;
    v16bf a = load_a(&sLeftT[mt * 16][0], LB, lane);
    v16bf b = load_b(&sQt[nt * 16][0], LB, lane);
    v8f acc = {};
    acc = wmma_bf16(a, b, acc);
    store_c(&LQ[((size_t)blk * LB + mt * 16) * DDIM + nt * 16], DDIM, acc, lane);
  }
  __syncthreads();

  if (tid < LB) {
    int l = tid;
    float eta = softplusf(step_size[(h * 2 + 0) * NSTEPS + step]);
    float invn = sInv[l];
    float proj = 0.f;
    for (int k = 0; k < LB; ++k) {
      float lsv = sLS[l][k];
      float dl = (sR2[k] * lsv * lsv - sT2[l][k]) * 2.f * lsv;
      proj += lsv * dl;
    }
    for (int k = 0; k < LB; ++k) {
      float lsv = sLS[l][k];
      float dl = (sR2[k] * lsv * lsv - sT2[l][k]) * 2.f * lsv;
      dl = (dl - lsv * proj) * invn;
      lpj[l * LB + k] = sLP[l][k] - eta * dl;
    }
  }
}

// Per (bh,k): t2r = LQ @ K^T; dr with projection; rp -= eta_r * dr.
__global__ void k_right_step(float* __restrict__ rp,
                             const float* __restrict__ keyg,
                             const float* __restrict__ LQ,
                             const float* __restrict__ l2w,
                             const float* __restrict__ step_size, int step) {
  int blk = blockIdx.x;  // bh*LB + k
  int bh = blk >> 5, k = blk & 31;
  int h = bh % NH;
  int tid = threadIdx.x, lane = tid & 31, wave = tid >> 5;
  __shared__ __align__(16) float  sRP[BSZ][BSZ];
  __shared__ __align__(16) float  sRS[BSZ][BSZ];
  __shared__ __align__(16) __bf16 sLQ[BSZ][DDIM];   // A: (j,v)
  __shared__ __align__(16) __bf16 sK[BSZ][DDIM];    // B (=K^T N-major): [i][v]
  __shared__ __align__(16) float  sT2[BSZ][BSZ];
  __shared__ float  sL2[BSZ];
  __shared__ float  sInv[BSZ];

  float* rpk = rp + (size_t)blk * BSZ * BSZ;
  const float* Kk = keyg + (size_t)blk * BSZ * DDIM;
  for (int idx = tid; idx < BSZ * BSZ / 4; idx += 256) {
    int e = 4 * idx;
    *(float4*)&sRP[e >> 6][e & 63] = *(const float4*)&rpk[e];
  }
  for (int idx = tid; idx < BSZ * DDIM / 4; idx += 256) {
    int e = 4 * idx;
    float4 k4 = *(const float4*)&Kk[e];
    v4bf pk = {(__bf16)k4.x, (__bf16)k4.y, (__bf16)k4.z, (__bf16)k4.w};
    *(v4bf*)&sK[e >> 6][e & 63] = pk;
  }
  for (int idx = tid; idx < BSZ * DDIM / 4; idx += 256) {
    int e = 4 * idx;
    int jj = e >> 6, v = e & 63;
    float4 q4 =
        *(const float4*)&LQ[(((size_t)bh * BSZ + jj) * LB + k) * DDIM + v];
    v4bf pk = {(__bf16)q4.x, (__bf16)q4.y, (__bf16)q4.z, (__bf16)q4.w};
    *(v4bf*)&sLQ[jj][v] = pk;
  }
  if (tid < BSZ) sL2[tid] = l2w[((size_t)bh * BSZ + tid) * LB + k];
  __syncthreads();

  if (tid < BSZ) {
    int jj = tid;
    const float4* row = (const float4*)&sRP[jj][0];
    float4* rowS = (float4*)&sRS[jj][0];
    float ss = 0.f;
    for (int i = 0; i < BSZ / 4; ++i) {
      float4 v = row[i];
      ss += v.x * v.x + v.y * v.y + v.z * v.z + v.w * v.w;
    }
    float n = sqrtf(ss);
    float inv = n > 0.f ? 1.f / n : 1.f;
    sInv[jj] = n > 0.f ? 1.f / n : 0.f;
    for (int i = 0; i < BSZ / 4; ++i) {
      float4 v = row[i];
      float4 s = {v.x * inv, v.y * inv, v.z * inv, v.w * inv};
      rowS[i] = s;
    }
  }
  __syncthreads();

#pragma unroll
  for (int t = wave; t < 16; t += 8) {  // t2r (64x64) = LQ @ K^T
    int mt = t >> 2, nt = t & 3;
    v8f acc = {};
#pragma unroll
    for (int ks = 0; ks < 2; ++ks) {
      v16bf a = load_a(&sLQ[mt * 16][ks * 32], DDIM, lane);
      v16bf b = load_b(&sK[nt * 16][ks * 32], DDIM, lane);
      acc = wmma_bf16(a, b, acc);
    }
    store_c(&sT2[mt * 16][nt * 16], BSZ, acc, lane);
  }
  __syncthreads();

  if (tid < BSZ) {
    int jj = tid;
    float eta = softplusf(step_size[(h * 2 + 1) * NSTEPS + step]);
    float l2v = sL2[jj];
    float invn = sInv[jj];
    float proj = 0.f;
    for (int i = 0; i < BSZ; ++i) {
      float rs = sRS[jj][i];
      float dr = (l2v * rs * rs - sT2[jj][i]) * 2.f * rs;
      proj += rs * dr;
    }
    for (int i = 0; i < BSZ; ++i) {
      float rs = sRS[jj][i];
      float dr = (l2v * rs * rs - sT2[jj][i]) * 2.f * rs;
      dr = (dr - rs * proj) * invn;
      rpk[jj * BSZ + i] = sRP[jj][i] - eta * dr;
    }
  }
}

// Per (bh,j): left_final = normalize(lp)^2; Z(32x64) = left @ Y_j, strided out.
__global__ void k_final_left(const float* __restrict__ lp,
                             const float* __restrict__ Y,
                             float* __restrict__ out) {
  int blk = blockIdx.x;  // bh*BSZ + j
  int bh = blk >> 6, j = blk & 63;
  int tid = threadIdx.x, lane = tid & 31, wave = tid >> 5;
  __shared__ __align__(16) float  sLP[LB][LB];
  __shared__ __align__(16) __bf16 sLeft[LB][LB];   // A: (l,k) row-major
  __shared__ __align__(16) __bf16 sYt[DDIM][LB];   // B N-major: [v][k]

  const float* lpj = lp + (size_t)blk * LB * LB;
  for (int idx = tid; idx < LB * LB / 4; idx += 256) {
    int e = 4 * idx;
    *(float4*)&sLP[e >> 5][e & 31] = *(const float4*)&lpj[e];
  }
  for (int idx = tid; idx < LB * DDIM / 4; idx += 256) {
    int e = 4 * idx;
    int k = e >> 6, v = e & 63;
    float4 y4 = *(const float4*)&Y[(((size_t)bh * LB + k) * BSZ + j) * DDIM + v];
    sYt[v + 0][k] = (__bf16)y4.x;
    sYt[v + 1][k] = (__bf16)y4.y;
    sYt[v + 2][k] = (__bf16)y4.z;
    sYt[v + 3][k] = (__bf16)y4.w;
  }
  __syncthreads();

  if (tid < LB) {
    int l = tid;
    const float4* row = (const float4*)&sLP[l][0];
    float ss = 0.f;
    for (int k = 0; k < LB / 4; ++k) {
      float4 v = row[k];
      ss += v.x * v.x + v.y * v.y + v.z * v.z + v.w * v.w;
    }
    float n = sqrtf(ss);
    float inv = n > 0.f ? 1.f / n : 1.f;
    for (int k = 0; k < LB / 4; ++k) {
      float4 v = row[k];
      float s0 = v.x * inv, s1 = v.y * inv, s2 = v.z * inv, s3 = v.w * inv;
      v4bf pk = {(__bf16)(s0 * s0), (__bf16)(s1 * s1), (__bf16)(s2 * s2),
                 (__bf16)(s3 * s3)};
      *(v4bf*)&sLeft[l][4 * k] = pk;
    }
  }
  __syncthreads();

  {  // 2x4 tiles over (l, v), one per wave
    int mt = wave >> 2, nt = wave & 3;
    v16bf a = load_a(&sLeft[mt * 16][0], LB, lane);
    v16bf b = load_b(&sYt[nt * 16][0], LB, lane);
    v8f acc = {};
    acc = wmma_bf16(a, b, acc);
    // out[(bh*SEQ + l*BSZ + j)*DDIM + v]; row stride in l is BSZ*DDIM
    float* dst =
        out + ((size_t)bh * SEQ + (size_t)(mt * 16) * BSZ + j) * DDIM + nt * 16;
    store_c(dst, BSZ * DDIM, acc, lane);
  }
}

// ---------------------------------------------------------------------------
extern "C" void kernel_launch(void* const* d_in, const int* in_sizes, int n_in,
                              void* d_out, int out_size, void* d_ws,
                              size_t ws_size, hipStream_t stream) {
  const float* qg     = (const float*)d_in[0];
  const float* keyg   = (const float*)d_in[1];
  const float* valg   = (const float*)d_in[2];
  const float* ascale = (const float*)d_in[3];
  const float* ssize  = (const float*)d_in[4];
  float* out = (float*)d_out;

  float* ws = (float*)d_ws;
  size_t off = 0;
  float* lp = ws + off; off += (size_t)NBH * BSZ * LB * LB;   // 3.1M floats
  float* rp = ws + off; off += (size_t)NBH * LB * BSZ * BSZ;  // 6.3M
  float* KR = ws + off; off += (size_t)NBH * LB * BSZ * DDIM; // 6.3M (reused as Y)
  float* LQ = ws + off; off += (size_t)NBH * BSZ * LB * DDIM; // 6.3M
  float* r2 = ws + off; off += (size_t)NBH * LB * BSZ;
  float* l2 = ws + off; off += (size_t)NBH * BSZ * LB;
  (void)in_sizes; (void)n_in; (void)out_size; (void)ws_size;

  {
    size_t nrp = (size_t)NBH * LB * BSZ * BSZ;
    int blocks = (int)((nrp + 255) / 256);
    k_init<<<blocks, 256, 0, stream>>>(lp, rp);
  }
  for (int s = 0; s < NSTEPS; ++s) {
    k_right_gemm<<<NBH * LB, 256, 0, stream>>>(rp, keyg, KR, r2);
    k_left_step<<<NBH * BSZ, 256, 0, stream>>>(lp, qg, KR, r2, ascale, ssize, s,
                                               LQ, l2);
    k_right_step<<<NBH * LB, 256, 0, stream>>>(rp, keyg, LQ, l2, ssize, s);
  }
  // Y = right_final @ V (into KR buffer), then Z = left_final @ Y
  k_right_gemm<<<NBH * LB, 256, 0, stream>>>(rp, valg, KR, (float*)nullptr);
  k_final_left<<<NBH * BSZ, 256, 0, stream>>>(lp, KR, out);
}
